// origin_ID_LSTM_86535001079967
// MI455X (gfx1250) — compile-verified, matrix operand order checked by reading, not run on hardware
//
#include <hip/hip_runtime.h>
#include <hip/hip_bf16.h>
#include <math.h>

// ---------------------------------------------------------------------------
// ID-LSTM for MI455X (gfx1250, wave32)
//   Phase 1: pad/convert X, W_ih, W_hh to bf16
//   Phase 2: Gx = X @ W_ih^T via v_wmma_f32_16x16x32_bf16 (write-bound)
//   Phase 3: recurrence distributed over 16 WGPs. Each workgroup owns 32
//            hidden units; its 128 W_hh gate rows (128KB bf16) stay RESIDENT
//            IN LDS. Policy decisions are computed redundantly+deterministically
//            per WGP (no sync on skipped steps); kept steps publish h/c to
//            parity-double-buffered global state and cross one device barrier.
// ---------------------------------------------------------------------------

typedef __attribute__((ext_vector_type(16))) __bf16 v16bf;
typedef __attribute__((ext_vector_type(8)))  __bf16 v8bf;
typedef __attribute__((ext_vector_type(8)))  float  v8f;

#define SEQLEN 32768
#define INPUTD 300
#define KPAD   320
#define HID    512
#define GATES  2048     // 4*HID
#define NWG    16       // workgroups (WGPs) cooperating on the recurrence
#define HSLICE 32       // hidden units per workgroup
#define RSLICE 128      // gate rows per workgroup (4*HSLICE)
#define STATE  1624     // 2*HID + 2*INPUTD

// ---------------- Phase 1: f32 -> bf16 with column zero-padding -------------
__global__ void cvt_pad_bf16(const float* __restrict__ src, __bf16* __restrict__ dst,
                             int rows, int cols, int cols_pad) {
  long long idx = (long long)blockIdx.x * blockDim.x + threadIdx.x;
  long long total = (long long)rows * cols_pad;
  if (idx >= total) return;
  int r = (int)(idx / cols_pad), c = (int)(idx % cols_pad);
  float v = (c < cols) ? src[(long long)r * cols + c] : 0.f;
  dst[idx] = (__bf16)v;
}

// ---------------- Phase 2: Gx[L,2048] = Xb[L,320] @ Wb[2048,320]^T ----------
__global__ void __launch_bounds__(256)
gemm_xwih_wmma(const __bf16* __restrict__ Xb, const __bf16* __restrict__ Wb,
               float* __restrict__ Gx) {
  const int wave = threadIdx.x >> 5;
  const int lane = threadIdx.x & 31;
  const int tile = blockIdx.x * 8 + wave;        // 2048*128 tiles total
  const int mt = tile >> 7;
  const int nt = tile & 127;
  const int mrow = mt * 16, nrow = nt * 16;

  const int half = lane >> 4;                    // K-subset select per ISA layout
  const int r    = lane & 15;
  const __bf16* arow = Xb + (size_t)(mrow + r) * KPAD + half * 8;
  const __bf16* brow = Wb + (size_t)(nrow + r) * KPAD + half * 8;

  v8f acc = {};
#pragma unroll
  for (int kk = 0; kk < KPAD; kk += 32) {
    v8bf alo = *(const v8bf*)(arow + kk);
    v8bf ahi = *(const v8bf*)(arow + kk + 16);
    v8bf blo = *(const v8bf*)(brow + kk);
    v8bf bhi = *(const v8bf*)(brow + kk + 16);
    v16bf a, b;
#pragma unroll
    for (int i = 0; i < 8; ++i) { a[i] = alo[i]; a[8 + i] = ahi[i];
                                  b[i] = blo[i]; b[8 + i] = bhi[i]; }
    acc = __builtin_amdgcn_wmma_f32_16x16x32_bf16(
        false, a, false, b, (short)0, acc, false, false);
  }
  const int n  = lane & 15;
  const int mb = (lane >> 4) * 8;
  float* out = Gx + (size_t)(mrow + mb) * GATES + nrow + n;
#pragma unroll
  for (int v = 0; v < 8; ++v) out[(size_t)v * GATES] = acc[v];
}

// ---------------- Phase 3: distributed sequential scan ----------------------
__device__ __forceinline__ float sigm(float x) { return 1.f / (1.f + __expf(-x)); }

__global__ void init_sync(int* p) { if (threadIdx.x < 2) p[threadIdx.x] = 0; }

__global__ void __launch_bounds__(512)
idlstm_scan(const float* __restrict__ X,          // [L,300] f32
            const long long* __restrict__ aspect, // [1]
            const float* __restrict__ AE_w,       // [128,300]
            const __bf16* __restrict__ Whh,       // [2048,512] bf16
            const float* __restrict__ b_ih,
            const float* __restrict__ b_hh,
            const float* __restrict__ Gx,         // [L,2048] f32
            const float* __restrict__ dec_W,      // [3,812]
            const float* __restrict__ dec_b,      // [3]
            const float* __restrict__ pnet_W,     // [2,1624]
            const float* __restrict__ pnet_b,     // [2]
            int*   __restrict__ bar,              // [2]: count, sense
            float* __restrict__ hg,               // [2][512] parity-buffered h
            float* __restrict__ cg,               // [2][512] parity-buffered c
            float* __restrict__ out)              // [4]
{
  __shared__ __bf16 sh_w[RSLICE * HID];   // 128KB: this WGP's W_hh gate rows
  __shared__ float  sh_h[HID], sh_c[HID]; // full-state mirrors
  __shared__ float  sh_pw[2 * STATE];     // policy weights (13KB)
  __shared__ float  sh_a[INPUTD];
  __shared__ float  sh_bias[RSLICE];
  __shared__ float  sh_g[RSLICE];
  __shared__ float  red0[16], red1[16];
  __shared__ int    sh_keep;

  const int tid  = threadIdx.x;           // 512 threads = 16 wave32
  const int lane = tid & 31;
  const int wid  = tid >> 5;
  const int wg   = blockIdx.x;            // 0..15
  const int wg32 = wg * HSLICE;

  // ---- one-time staging ----
  sh_h[tid] = 0.f; sh_c[tid] = 0.f;       // tid covers exactly HID=512
  for (int i = tid; i < INPUTD; i += 512) sh_a[i] = AE_w[(int)aspect[0] * INPUTD + i];
  for (int i = tid; i < 2 * STATE; i += 512) sh_pw[i] = pnet_W[i];
  if (tid < RSLICE) {
    int grow = (tid >> 5) * HID + wg32 + (tid & 31);
    sh_bias[tid] = b_ih[grow] + b_hh[grow];
  }
  // W_hh slice -> LDS (16 waves x 8 rows each; v8bf = 16B moves)
  for (int r = wid; r < RSLICE; r += 16) {
    int grow = (r >> 5) * HID + wg32 + (r & 31);
    const __bf16* src = Whh + (size_t)grow * HID;
#pragma unroll
    for (int k = lane * 8; k < HID; k += 32 * 8)
      *(v8bf*)(sh_w + r * HID + k) = *(const v8bf*)(src + k);
  }
  __syncthreads();

  const float pb0 = pnet_b[0], pb1 = pnet_b[1];
  int keptCount = 0, lastPar = 0, dirty = 0, lsense = 0;

  for (int t = 0; t < SEQLEN; ++t) {
    // refresh full-state mirror after a kept step (publishes already fenced+barriered)
    if (dirty) {
      __threadfence();
      volatile const float* vh = hg + lastPar * HID;
      volatile const float* vc = cg + lastPar * HID;
      sh_h[tid] = vh[tid];
      sh_c[tid] = vc[tid];
      __syncthreads();
      dirty = 0;
    }
    if (t + 1 < SEQLEN && tid < 8)
      __builtin_prefetch(X + (size_t)(t + 1) * INPUTD + tid * 40, 0, 1);

    // ---- policy: logits = [c,h,x,a] . pnet_W^T + pnet_b (identical per WGP) ----
    float p0 = 0.f, p1 = 0.f;
    {
      float sv = sh_c[tid];
      p0 += sv * sh_pw[tid];            p1 += sv * sh_pw[STATE + tid];
      sv = sh_h[tid];
      p0 += sv * sh_pw[512 + tid];      p1 += sv * sh_pw[STATE + 512 + tid];
      int idx = tid + 1024;
      if (idx < STATE) {
        sv = (idx < 1324) ? X[(size_t)t * INPUTD + (idx - 1024)] : sh_a[idx - 1324];
        p0 += sv * sh_pw[idx];          p1 += sv * sh_pw[STATE + idx];
      }
    }
#pragma unroll
    for (int o = 16; o; o >>= 1) {
      p0 += __shfl_down(p0, o, 32);
      p1 += __shfl_down(p1, o, 32);
    }
    if (lane == 0) { red0[wid] = p0; red1[wid] = p1; }
    __syncthreads();
    if (tid == 0) {
      float l0 = pb0, l1 = pb1;
#pragma unroll
      for (int w = 0; w < 16; ++w) { l0 += red0[w]; l1 += red1[w]; }
      unsigned long long z = (0x9E3779B97F4A7C15ull * (unsigned long long)(t + 1))
                             ^ 0x853C49E6748FEA9Bull;
      z ^= z >> 30; z *= 0xBF58476D1CE4E5B9ull;
      z ^= z >> 27; z *= 0x94D049BB133111EBull;
      z ^= z >> 31;
      float u = (float)(z >> 40) * (1.f / 16777216.f);
      sh_keep = (u < sigm(l1 - l0)) ? 1 : 0;
    }
    __syncthreads();
    const int keep = sh_keep;           // uniform across block AND across WGPs
    __syncthreads();

    if (keep) {
      // ---- gate matvec from LDS-resident weights: 4 threads per row ----
      const int r = tid >> 2;           // 0..127 local gate row
      const int q = tid & 3;            // k-quarter
      float acc = 0.f;
      const __bf16* wr = sh_w + r * HID + q * 128;
      const float*  hh = sh_h + q * 128;
      for (int k = 0; k < 128; k += 8) {
        v8bf wv = *(const v8bf*)(wr + k);      // ds_load_b128
#pragma unroll
        for (int j = 0; j < 8; ++j) acc = fmaf((float)wv[j], hh[k + j], acc);
      }
      acc += __shfl_down(acc, 2, 4);
      acc += __shfl_down(acc, 1, 4);
      if (q == 0) {
        int grow = (r >> 5) * HID + wg32 + (r & 31);
        sh_g[r] = acc + Gx[(size_t)t * GATES + grow] + sh_bias[r];
      }
      __syncthreads();

      const int par = keptCount & 1;    // publish parity (uniform across WGPs)
      if (tid < HSLICE) {
        const int j = tid;
        float gi = sh_g[j], gf = sh_g[32 + j], gg = sh_g[64 + j], go = sh_g[96 + j];
        float cc = sh_c[wg32 + j];
        float c2 = sigm(gf) * cc + sigm(gi) * tanhf(gg);
        float h2 = sigm(go) * tanhf(c2);
        hg[par * HID + wg32 + j] = h2;  // publish this WGP's 32-unit slice
        cg[par * HID + wg32 + j] = c2;
      }
      __threadfence();                  // release publishes to device scope
      __syncthreads();

      // ---- device-scope sense-reversing barrier across the 16 WGPs ----
      if (tid == 0) {
        lsense ^= 1;
        int prev = atomicAdd(&bar[0], 1);
        if (prev == NWG - 1) {
          atomicExch(&bar[0], 0);
          __threadfence();
          atomicExch(&bar[1], lsense);
        } else {
          while (atomicAdd(&bar[1], 0) != lsense) __builtin_amdgcn_s_sleep(2);
        }
      }
      __syncthreads();

      lastPar = par;
      ++keptCount;
      dirty = 1;
    }
  }

  // ---- epilogue: WGP 0 decodes ----
  if (dirty) {
    __threadfence();
    volatile const float* vh = hg + lastPar * HID;
    sh_h[tid] = vh[tid];
    __syncthreads();
  }
  if (wg == 0) {
    if (tid < 3) {
      float acc = dec_b[tid];
      const float* w = dec_W + tid * (HID + INPUTD);
      for (int k = 0; k < HID; ++k)    acc = fmaf(w[k],       sh_h[k], acc);
      for (int k = 0; k < INPUTD; ++k) acc = fmaf(w[HID + k], sh_a[k], acc);
      out[tid] = acc;
    }
    if (tid == 3) out[3] = (float)keptCount;
  }
}

// ---------------------------------------------------------------------------
extern "C" void kernel_launch(void* const* d_in, const int* in_sizes, int n_in,
                              void* d_out, int out_size, void* d_ws, size_t ws_size,
                              hipStream_t stream) {
  const float*     X      = (const float*)d_in[0];
  const long long* aspect = (const long long*)d_in[1];
  const float*     AE_w   = (const float*)d_in[2];
  const float*     W_ih   = (const float*)d_in[3];
  const float*     W_hh   = (const float*)d_in[4];
  const float*     b_ih   = (const float*)d_in[5];
  const float*     b_hh   = (const float*)d_in[6];
  const float*     dec_W  = (const float*)d_in[7];
  const float*     dec_b  = (const float*)d_in[8];
  const float*     pnet_W = (const float*)d_in[9];
  const float*     pnet_b = (const float*)d_in[10];

  // workspace layout
  char*  ws   = (char*)d_ws;
  int*   bar  = (int*)ws;                                  // 256B sync area
  float* hg   = (float*)(ws + 256);                        // [2][512]
  float* cg   = (float*)(ws + 256 + 2 * HID * 4);          // [2][512]
  char*  big  = ws + 16384;
  __bf16* Xb  = (__bf16*)big;
  size_t  oXb = (size_t)SEQLEN * KPAD * sizeof(__bf16);
  __bf16* Wb  = (__bf16*)(big + oXb);
  size_t  oWb = (size_t)GATES * KPAD * sizeof(__bf16);
  __bf16* Whh = (__bf16*)(big + oXb + oWb);
  size_t  oWh = (size_t)GATES * HID * sizeof(__bf16);
  float*  Gx  = (float*)(big + oXb + oWb + oWh);

  init_sync<<<1, 64, 0, stream>>>(bar);

  {
    long long total = (long long)SEQLEN * KPAD;
    cvt_pad_bf16<<<(unsigned)((total + 255) / 256), 256, 0, stream>>>(X, Xb, SEQLEN, INPUTD, KPAD);
    total = (long long)GATES * KPAD;
    cvt_pad_bf16<<<(unsigned)((total + 255) / 256), 256, 0, stream>>>(W_ih, Wb, GATES, INPUTD, KPAD);
    total = (long long)GATES * HID;
    cvt_pad_bf16<<<(unsigned)((total + 255) / 256), 256, 0, stream>>>(W_hh, Whh, GATES, HID, HID);
  }

  {
    const int tiles = (SEQLEN / 16) * (GATES / 16);
    gemm_xwih_wmma<<<tiles / 8, 256, 0, stream>>>(Xb, Wb, Gx);
  }

  idlstm_scan<<<NWG, 512, 0, stream>>>(X, aspect, AE_w, Whh, b_ih, b_hh, Gx,
                                       dec_W, dec_b, pnet_W, pnet_b,
                                       bar, hg, cg, (float*)d_out);
}
